// PatchedDeepseekAttention_55894704390471
// MI455X (gfx1250) — compile-verified, matrix operand order checked by reading, not run on hardware
//
#include <hip/hip_runtime.h>
#include <hip/hip_bf16.h>

// ---------------------------------------------------------------------------
// MI455X (gfx1250) fused attention block.
// B=8 S=1024 HID=4096 H=32 HKV=8 D=128. Precision: bf16 WMMA, f32 accum.
// The paged-KV scatter/gather in the reference is an identity on the fresh
// K/V (history==0, zero-initialized cache, same permutation both ways), so
// the cache buffers are skipped entirely.
//
// All WMMA operands are pre-packed into CDNA5 fragment layout so every
// operand fetch in a hot loop is a single coalesced 32B/lane b128 load.
// ---------------------------------------------------------------------------

typedef __attribute__((ext_vector_type(16))) __bf16 v16bf;
typedef __attribute__((ext_vector_type(8)))  float  v8f;

#define TOTAL   8192
#define HID_    4096
#define NH      32
#define NKV     8
#define HD      128
#define SLEN    1024

__device__ __forceinline__ __bf16 f2bf(float f) { return (__bf16)f; }

// CDNA5 16-bit A/B fragment K index for element e (0..15) in lane-half g.
// ISA 7.12.2: VGPR0..3 hold K {0..7 | 8..15}, VGPR4..7 hold K {16..23 | 24..31}.
__device__ __forceinline__ int kmap(int e, int g) {
    int v = e >> 1, h = e & 1;
    return (v < 4) ? (v * 2 + h + g * 8) : (16 + (v - 4) * 2 + h + g * 8);
}

__device__ __forceinline__ v8f wmma_bf16(v16bf a, v16bf b, v8f c) {
    return __builtin_amdgcn_wmma_f32_16x16x32_bf16(
        /*neg_a=*/false, a, /*neg_b=*/false, b,
        /*c_mod=*/(short)0, c, /*reuse_a=*/false, /*reuse_b=*/false);
}

// ---------------------------------------------------------------------------
// Pack fp32 row-major A[M,K] into bf16 A-fragments (16x32 tiles), one wave
// per fragment, 32B/lane stores. frag(mt,kt) at ((mt*Kt + kt)*512).
// ---------------------------------------------------------------------------
__global__ __launch_bounds__(256) void pack_a_kernel(const float* __restrict__ A,
                                                     __bf16* __restrict__ Ap,
                                                     int M, int K) {
    int frag = (blockIdx.x * blockDim.x + threadIdx.x) >> 5;
    int lane = threadIdx.x & 31;
    int Kt = K >> 5;
    int mt = frag / Kt, kt = frag - mt * Kt;
    if (mt >= (M >> 4)) return;
    int m = mt * 16 + (lane & 15);
    int g = lane >> 4;
    const float* row = A + (size_t)m * K + kt * 32;
    v16bf out;
#pragma unroll
    for (int v = 0; v < 8; ++v) {                 // k pairs are contiguous
        int k = kmap(2 * v, g);
        float2 w = *(const float2*)(row + k);
        out[2 * v]     = f2bf(w.x);
        out[2 * v + 1] = f2bf(w.y);
    }
    *(v16bf*)(Ap + (size_t)frag * 512 + lane * 16) = out;
}

// ---------------------------------------------------------------------------
// Pack fp32 row-major B[K,N] into bf16 B-fragments (32x16 tiles).
// frag(nt,kt) at ((nt*Kt + kt)*512); lane carries column n = lane&15.
// ---------------------------------------------------------------------------
__global__ __launch_bounds__(256) void pack_b_kernel(const float* __restrict__ B,
                                                     __bf16* __restrict__ Bp,
                                                     int K, int N) {
    int frag = (blockIdx.x * blockDim.x + threadIdx.x) >> 5;
    int lane = threadIdx.x & 31;
    int Kt = K >> 5;
    int nt = frag / Kt, kt = frag - nt * Kt;
    if (nt >= (N >> 4)) return;
    int n = nt * 16 + (lane & 15);
    int g = lane >> 4;
    v16bf out;
#pragma unroll
    for (int e = 0; e < 16; ++e) {
        int k = kt * 32 + kmap(e, g);
        out[e] = f2bf(B[(size_t)k * N + n]);
    }
    *(v16bf*)(Bp + (size_t)frag * 512 + lane * 16) = out;
}

// ---------------------------------------------------------------------------
// Pack RoPE'd K (fp32 [TOTAL, NKV, HD]) into B-fragments of K^T for QK^T:
// frag((b*NKV+kvh), nt = s-tile 0..63, kt = d-chunk 0..3); lane n = seq pos,
// element k = d. d pairs contiguous -> float2 loads.
// ---------------------------------------------------------------------------
__global__ __launch_bounds__(256) void pack_k_frag(const float* __restrict__ Kf,
                                                   __bf16* __restrict__ Kp) {
    int frag = (blockIdx.x * blockDim.x + threadIdx.x) >> 5;
    int lane = threadIdx.x & 31;
    if (frag >= 8 * NKV * 64 * 4) return;
    int kt  = frag & 3;
    int nt  = (frag >> 2) & 63;
    int bk  = frag >> 8;              // b*NKV + kvh
    int b   = bk >> 3, kvh = bk & 7;
    int g = lane >> 4, ln = lane & 15;
    int s = nt * 16 + ln;
    const float* row = Kf + (size_t)(b * SLEN + s) * (NKV * HD) + kvh * HD + kt * 32;
    v16bf out;
#pragma unroll
    for (int v = 0; v < 8; ++v) {
        int d = kmap(2 * v, g);
        float2 w = *(const float2*)(row + d);
        out[2 * v]     = f2bf(w.x);
        out[2 * v + 1] = f2bf(w.y);
    }
    *(v16bf*)(Kp + (size_t)frag * 512 + lane * 16) = out;
}

// ---------------------------------------------------------------------------
// Pack V (fp32 [TOTAL, NKV, HD]) into B-fragments for PxV:
// frag((b*NKV+kvh), ct = s-chunk 0..31, j = d-tile 0..7); lane n = d,
// element k = seq pos within chunk. Lane-coalesced reads (n = ln).
// ---------------------------------------------------------------------------
__global__ __launch_bounds__(256) void pack_v_frag(const float* __restrict__ Vf,
                                                   __bf16* __restrict__ Vp) {
    int frag = (blockIdx.x * blockDim.x + threadIdx.x) >> 5;
    int lane = threadIdx.x & 31;
    if (frag >= 8 * NKV * 32 * 8) return;
    int j   = frag & 7;
    int ct  = (frag >> 3) & 31;
    int bk  = frag >> 8;              // b*NKV + kvh
    int b   = bk >> 3, kvh = bk & 7;
    int g = lane >> 4, ln = lane & 15;
    int d = j * 16 + ln;
    v16bf out;
#pragma unroll
    for (int e = 0; e < 16; ++e) {
        int s = ct * 32 + kmap(e, g);
        out[e] = f2bf(Vf[(size_t)(b * SLEN + s) * (NKV * HD) + kvh * HD + d]);
    }
    *(v16bf*)(Vp + (size_t)frag * 512 + lane * 16) = out;
}

// ---------------------------------------------------------------------------
// GEMM: C[M,N] = Apacked x Bpacked. One wave owns a 64x64 C tile:
// 16 f32 accumulator fragments, 16 wmma per K-step of 32.
// ---------------------------------------------------------------------------
__global__ __launch_bounds__(256) void gemm_bf16_kernel(const __bf16* __restrict__ Ap,
                                                        const __bf16* __restrict__ Bp,
                                                        float* __restrict__ C,
                                                        int M, int N, int K) {
    int gw = (blockIdx.x * blockDim.x + threadIdx.x) >> 5;
    int lane = threadIdx.x & 31;
    int tilesN = N >> 6;
    int tilesM = M >> 6;
    if (gw >= tilesM * tilesN) return;
    int tm = gw / tilesN, tn = gw - tm * tilesN;
    int Kt = K >> 5;
    int g = lane >> 4, ln = lane & 15;

    v8f acc[4][4] = {};
    const size_t laneOff = (size_t)lane * 16;
    for (int kt = 0; kt < Kt; ++kt) {
        v16bf a[4], b[4];
#pragma unroll
        for (int i = 0; i < 4; ++i)
            a[i] = *(const v16bf*)(Ap + ((size_t)(tm * 4 + i) * Kt + kt) * 512 + laneOff);
#pragma unroll
        for (int j = 0; j < 4; ++j)
            b[j] = *(const v16bf*)(Bp + ((size_t)(tn * 4 + j) * Kt + kt) * 512 + laneOff);
#pragma unroll
        for (int i = 0; i < 4; ++i)
#pragma unroll
            for (int j = 0; j < 4; ++j)
                acc[i][j] = wmma_bf16(a[i], b[j], acc[i][j]);
    }
#pragma unroll
    for (int i = 0; i < 4; ++i)
#pragma unroll
        for (int j = 0; j < 4; ++j) {
            int n = tn * 64 + j * 16 + ln;
#pragma unroll
            for (int v = 0; v < 8; ++v) {
                int m = tm * 64 + i * 16 + v + g * 8;
                C[(size_t)m * N + n] = acc[i][j][v];
            }
        }
}

// ---------------------------------------------------------------------------
// RoPE in-place on fp32 [TOTAL, nheads, 128]; pos = token % SLEN.
// ---------------------------------------------------------------------------
__global__ __launch_bounds__(256) void rope_kernel(float* __restrict__ x, int nheads) {
    int idx = blockIdx.x * blockDim.x + threadIdx.x;
    int n = TOTAL * nheads * 64;
    if (idx >= n) return;
    int d = idx & 63;
    int hh = (idx >> 6) % nheads;
    int t = idx / (64 * nheads);
    int pos = t & (SLEN - 1);
    float inv = __powf(10000.0f, -(float)d / 64.0f);
    float ang = (float)pos * inv;
    float c = __cosf(ang), s = __sinf(ang);
    float* p = x + ((size_t)t * nheads + hh) * HD;
    float x1 = p[d], x2 = p[d + 64];
    p[d]      = x1 * c - x2 * s;
    p[d + 64] = x2 * c + x1 * s;
}

// ---------------------------------------------------------------------------
// Flash attention: one wave per (batch, head, 16-row query tile).
// Per 32-wide KV chunk: 8 K-fragment loads + 8 wmma (QK^T), LDS relayout of
// P (C->A layout), 8 V-fragment loads + 8 wmma (PV). All fragment loads are
// coalesced b128. Online softmax with per-lane row stats, cross-lane row
// reductions via shfl_xor within 16-lane halves.
// ---------------------------------------------------------------------------
__global__ __launch_bounds__(128) void attn_kernel(const float*  __restrict__ Qf,
                                                   const __bf16* __restrict__ Kp,
                                                   const __bf16* __restrict__ Vp,
                                                   float* __restrict__ Of) {
    const int lane = threadIdx.x & 31;
    const int wib  = threadIdx.x >> 5;
    const int gw   = blockIdx.x * 4 + wib;
    const int qt   = gw & 63;            // query tile within sequence (64 tiles)
    const int h    = (gw >> 6) & 31;     // head
    const int b    = gw >> 11;           // batch
    const int kvh  = h >> 2;             // GQA: 4 query heads per kv head
    const int g    = lane >> 4;
    const int ln   = lane & 15;
    __shared__ float ldsP[4][512];       // per-wave 16x32 P staging
    float* myP = ldsP[wib];

    const float scale = 0.08838834764831845f;   // 1/sqrt(128)
    const int q0 = b * SLEN + qt * 16;
    const size_t laneOff = (size_t)lane * 16;
    const __bf16* kbase = Kp + (size_t)(b * NKV + kvh) * 64 * 4 * 512;
    const __bf16* vbase = Vp + (size_t)(b * NKV + kvh) * 32 * 8 * 512;

    // Q A-fragments (16x32 over D, 4 chunks) resident in VGPRs (one-time gather).
    v16bf aq[4];
#pragma unroll
    for (int dk = 0; dk < 4; ++dk) {
        const float* qrow = Qf + (size_t)(q0 + ln) * (NH * HD) + h * HD + dk * 32;
#pragma unroll
        for (int v = 0; v < 8; ++v) {
            int d = kmap(2 * v, g);
            float2 w = *(const float2*)(qrow + d);
            aq[dk][2 * v]     = f2bf(w.x);
            aq[dk][2 * v + 1] = f2bf(w.y);
        }
    }

    v8f ov[8] = {};
    float rm[8], rl[8];
#pragma unroll
    for (int v = 0; v < 8; ++v) { rm[v] = -3.0e38f; rl[v] = 0.0f; }

    const int nch = (qt * 16 + 16 + 31) >> 5;   // causal chunk count
    for (int ci = 0; ci < nch; ++ci) {
        const int sbase = ci * 32;
        // ---- scores 16x32 = Q(16x128) x K^T(128x32), packed K fragments ----
        v8f s0 = {}, s1 = {};
#pragma unroll
        for (int dk = 0; dk < 4; ++dk) {
            v16bf bk0 = *(const v16bf*)(kbase + ((size_t)(ci * 2 + 0) * 4 + dk) * 512 + laneOff);
            v16bf bk1 = *(const v16bf*)(kbase + ((size_t)(ci * 2 + 1) * 4 + dk) * 512 + laneOff);
            s0 = wmma_bf16(aq[dk], bk0, s0);
            s1 = wmma_bf16(aq[dk], bk1, s1);
        }
        // ---- scale + causal mask + row max (reduce over 16 lanes/half) ----
        float cm[8];
#pragma unroll
        for (int v = 0; v < 8; ++v) {
            int qpos = qt * 16 + v + g * 8;
            float e0 = s0[v] * scale, e1 = s1[v] * scale;
            if (sbase + ln > qpos)      e0 = -1e30f;
            if (sbase + 16 + ln > qpos) e1 = -1e30f;
            s0[v] = e0; s1[v] = e1;
            cm[v] = fmaxf(e0, e1);
        }
#pragma unroll
        for (int v = 0; v < 8; ++v)
#pragma unroll
            for (int off = 8; off >= 1; off >>= 1)
                cm[v] = fmaxf(cm[v], __shfl_xor(cm[v], off, 32));
        // ---- online softmax update ----
        float fl[8];
#pragma unroll
        for (int v = 0; v < 8; ++v) {
            float nm = fmaxf(rm[v], cm[v]);
            fl[v] = __expf(rm[v] - nm);
            rm[v] = nm;
            float p0 = __expf(s0[v] - nm), p1 = __expf(s1[v] - nm);
            s0[v] = p0; s1[v] = p1;
            float cs = p0 + p1;
#pragma unroll
            for (int off = 8; off >= 1; off >>= 1)
                cs += __shfl_xor(cs, off, 32);
            rl[v] = rl[v] * fl[v] + cs;
        }
#pragma unroll
        for (int j = 0; j < 8; ++j)
#pragma unroll
            for (int v = 0; v < 8; ++v) ov[j][v] *= fl[v];
        // ---- P: C-fragment layout -> LDS -> A-fragment layout ----
#pragma unroll
        for (int v = 0; v < 8; ++v) {
            int m = v + g * 8;
            myP[m * 32 + ln]      = s0[v];
            myP[m * 32 + 16 + ln] = s1[v];
        }
        asm volatile("s_wait_dscnt 0x0" ::: "memory");
        v16bf pa;
#pragma unroll
        for (int e = 0; e < 16; ++e)
            pa[e] = f2bf(myP[ln * 32 + kmap(e, g)]);
        // ---- O += P x V, 8 d-tiles of 16, packed V fragments ----
#pragma unroll
        for (int j = 0; j < 8; ++j) {
            v16bf bv = *(const v16bf*)(vbase + ((size_t)ci * 8 + j) * 512 + laneOff);
            ov[j] = wmma_bf16(pa, bv, ov[j]);
        }
    }
    // ---- normalize + store ----
#pragma unroll
    for (int j = 0; j < 8; ++j)
#pragma unroll
        for (int v = 0; v < 8; ++v) {
            int m = v + g * 8;
            Of[(size_t)(q0 + m) * (NH * HD) + h * HD + j * 16 + ln] = ov[j][v] / rl[v];
        }
}

// ---------------------------------------------------------------------------
// Host-side orchestration. Workspace layout (464 MB), with aliasing:
//   [0,64MB)    hidden packed  -> (after QKV gemms) Kp(16MB)+Vp(16MB)
//               -> (after attention) attn packed
//   [64,96)     Wq packed   [96,104) Wk packed   [104,112) Wv packed
//   [112,144)   Wo packed
//   [144,272)   Q fp32      [272,304) K fp32     [304,336) V fp32
//   [336,464)   attn fp32
// All reuse is safe under stream ordering and repeats identically per call.
// ---------------------------------------------------------------------------
extern "C" void kernel_launch(void* const* d_in, const int* in_sizes, int n_in,
                              void* d_out, int out_size, void* d_ws, size_t ws_size,
                              hipStream_t stream) {
    (void)in_sizes; (void)n_in; (void)out_size;
    const float* hidden = (const float*)d_in[0];
    const float* Wq = (const float*)d_in[1];
    const float* Wk = (const float*)d_in[2];
    const float* Wv = (const float*)d_in[3];
    const float* Wo = (const float*)d_in[4];

    const size_t MB = 1ull << 20;
    if (ws_size < 464ull * MB) return;
    char* ws = (char*)d_ws;
    __bf16* hp  = (__bf16*)(ws);               // 64 MB: hidden/attn packed (reused)
    __bf16* Kpp = (__bf16*)(ws);               // 16 MB: K fragments (aliases hp)
    __bf16* Vpp = (__bf16*)(ws + 16  * MB);    // 16 MB: V fragments (aliases hp)
    __bf16* wqp = (__bf16*)(ws + 64  * MB);    // 32 MB
    __bf16* wkp = (__bf16*)(ws + 96  * MB);    //  8 MB
    __bf16* wvp = (__bf16*)(ws + 104 * MB);    //  8 MB
    __bf16* wop = (__bf16*)(ws + 112 * MB);    // 32 MB
    float*  Qb  = (float*)(ws + 144 * MB);     // 128 MB
    float*  Kb  = (float*)(ws + 272 * MB);     //  32 MB
    float*  Vb  = (float*)(ws + 304 * MB);     //  32 MB
    float*  Ab  = (float*)(ws + 336 * MB);     // 128 MB

    auto cdiv = [](long a, long b) { return (unsigned)((a + b - 1) / b); };

    // Pack weights (one-time) and activations.
    long fb_big = (HID_ / 16) * (HID_ / 32);          // 32768 frags (Wq, Wo)
    long fb_kv  = ((NKV * HD) / 16) * (HID_ / 32);    //  8192 frags (Wk, Wv)
    long fa     = (TOTAL / 16) * (HID_ / 32);         // 65536 frags
    pack_b_kernel<<<cdiv(fb_big * 32, 256), 256, 0, stream>>>(Wq, wqp, HID_, HID_);
    pack_b_kernel<<<cdiv(fb_kv  * 32, 256), 256, 0, stream>>>(Wk, wkp, HID_, NKV * HD);
    pack_b_kernel<<<cdiv(fb_kv  * 32, 256), 256, 0, stream>>>(Wv, wvp, HID_, NKV * HD);
    pack_b_kernel<<<cdiv(fb_big * 32, 256), 256, 0, stream>>>(Wo, wop, HID_, HID_);
    pack_a_kernel<<<cdiv(fa * 32, 256), 256, 0, stream>>>(hidden, hp, TOTAL, HID_);

    // QKV projections (hp dead afterwards).
    long wq_waves  = (TOTAL / 64) * (HID_ / 64);
    long wkv_waves = (TOTAL / 64) * ((NKV * HD) / 64);
    gemm_bf16_kernel<<<cdiv(wq_waves * 32, 256), 256, 0, stream>>>(hp, wqp, Qb, TOTAL, HID_, HID_);
    gemm_bf16_kernel<<<cdiv(wkv_waves * 32, 256), 256, 0, stream>>>(hp, wkp, Kb, TOTAL, NKV * HD, HID_);
    gemm_bf16_kernel<<<cdiv(wkv_waves * 32, 256), 256, 0, stream>>>(hp, wvp, Vb, TOTAL, NKV * HD, HID_);

    // RoPE.
    rope_kernel<<<cdiv((long)TOTAL * NH  * 64, 256), 256, 0, stream>>>(Qb, NH);
    rope_kernel<<<cdiv((long)TOTAL * NKV * 64, 256), 256, 0, stream>>>(Kb, NKV);

    // Pack K/V into WMMA B-fragment layout (into the dead hp region).
    long fk = 8L * NKV * 64 * 4;   // 16384 frags
    long fv = 8L * NKV * 32 * 8;   // 16384 frags
    pack_k_frag<<<cdiv(fk * 32, 256), 256, 0, stream>>>(Kb, Kpp);
    pack_v_frag<<<cdiv(fv * 32, 256), 256, 0, stream>>>(Vb, Vpp);

    // Flash attention: 8*32*64 = 16384 waves, 4 waves/block.
    attn_kernel<<<4096, 128, 0, stream>>>(Qb, Kpp, Vpp, Ab);

    // Output projection into d_out (fp32). hp reuse overwrites dead Kp/Vp.
    pack_a_kernel<<<cdiv(fa * 32, 256), 256, 0, stream>>>(Ab, hp, TOTAL, HID_);
    gemm_bf16_kernel<<<cdiv(wq_waves * 32, 256), 256, 0, stream>>>(hp, wop, (float*)d_out, TOTAL, HID_, HID_);
}
